// TheSonnetsRNN_48395691491606
// MI455X (gfx1250) — compile-verified, matrix operand order checked by reading, not run on hardware
//
#include <hip/hip_runtime.h>
#include <hip/hip_bf16.h>

// ---------------- types ----------------
typedef _Float16 v16h __attribute__((ext_vector_type(16)));
typedef float    v8f  __attribute__((ext_vector_type(8)));
typedef float    v2f  __attribute__((ext_vector_type(2)));

#define RNN_B 16
#define RNN_H 1024
#define RNN_V 32000
#define RNN_T 128

// ---------------- CDNA5 async copy helpers ----------------
// GLOBAL_LOAD_ASYNC_TO_LDS_B128: copies 16B from global memory to LDS at the offset in vdst, tracked
// with ASYNCcnt (CDNA5 ISA). GV addressing mode.
__device__ __forceinline__ void async_copy_b128(unsigned long long gaddr,
                                                unsigned int lds_off) {
    asm volatile("global_load_async_to_lds_b128 %0, %1, off"
                 :: "v"(lds_off), "v"(gaddr)
                 : "memory");
}
__device__ __forceinline__ void wait_async_le8() {
    asm volatile("s_wait_asynccnt 0x8" ::: "memory");
}
__device__ __forceinline__ void wait_async_le0() {
    asm volatile("s_wait_asynccnt 0x0" ::: "memory");
}
// Low 32 bits of a generic (shared-aperture) pointer = LDS byte offset.
__device__ __forceinline__ unsigned int lds_offset_of(const void* p) {
    return (unsigned int)(unsigned long long)p;
}

// =====================================================================
// Recurrent step GEMM:  out = act( A1@W1 [+ A2@W2] [+ bias] [+ addmat] )
// A: [16 x 1024] fp32, W: [1024 x 1024] fp32 row-major (L2-resident).
// Grid: 64 blocks (one 16x16 N-tile each), 256 threads = 8 waves.
// Each wave handles a 128-wide K slice with V_WMMA_F32_16X16X4_F32,
// partials reduced through LDS, epilogue fused. Latency-bound phase ->
// fp32 keeps the 128-step recurrence numerically safe at zero cost.
// =====================================================================
__global__ __launch_bounds__(256) void rnn_step_gemm(
    const float* __restrict__ A1, const float* __restrict__ W1,
    const float* __restrict__ A2, const float* __restrict__ W2,
    const float* __restrict__ bias, const float* __restrict__ addmat,
    float* __restrict__ out32, _Float16* __restrict__ out16, int do_tanh)
{
    __shared__ float red[8 * 256];

    const int tid  = threadIdx.x;
    const int lane = tid & 31;
    const int w    = tid >> 5;
    const int l16  = lane & 15;
    const int s    = lane >> 4;          // K-phase within wave (ISA A/B layout)
    const int ncol = blockIdx.x * 16 + l16;

    v8f acc = {};

    // ---- A1 @ W1 over K slice [w*128, w*128+128) ----
    {
        const int k0 = w * 128;
        for (int k = k0; k < k0 + 128; k += 4) {
            v2f a = *(const v2f*)(A1 + (size_t)l16 * RNN_H + k + s * 2);
            v2f b;
            const int rk = k + s * 2;
            b.x = W1[(size_t)rk * RNN_H + ncol];
            b.y = W1[(size_t)(rk + 1) * RNN_H + ncol];
            acc = __builtin_amdgcn_wmma_f32_16x16x4_f32(
                false, a, false, b, (short)0, acc, false, false);
        }
    }
    // ---- optional A2 @ W2 accumulated into same tile ----
    if (A2 != nullptr) {
        const int k0 = w * 128;
        for (int k = k0; k < k0 + 128; k += 4) {
            v2f a = *(const v2f*)(A2 + (size_t)l16 * RNN_H + k + s * 2);
            v2f b;
            const int rk = k + s * 2;
            b.x = W2[(size_t)rk * RNN_H + ncol];
            b.y = W2[(size_t)(rk + 1) * RNN_H + ncol];
            acc = __builtin_amdgcn_wmma_f32_16x16x4_f32(
                false, a, false, b, (short)0, acc, false, false);
        }
    }

    // ---- spill this wave's 16x16 tile into LDS (element id = m*16+n) ----
    #pragma unroll
    for (int i = 0; i < 8; ++i) {
        const int e = (i + s * 8) * 16 + l16;   // C/D layout: M = i + s*8, N = l16
        red[w * 256 + e] = acc[i];
    }
    __syncthreads();

    // ---- reduce 8 K-split partials; fused epilogue ----
    float v = 0.f;
    #pragma unroll
    for (int ww = 0; ww < 8; ++ww) v += red[ww * 256 + tid];

    const int m  = tid >> 4;
    const int nn = tid & 15;
    const int nc = blockIdx.x * 16 + nn;
    if (bias)   v += bias[nc];
    if (addmat) v += addmat[(size_t)m * RNN_H + nc];
    if (do_tanh) v = tanhf(v);
    out32[(size_t)m * RNN_H + nc] = v;
    if (out16) out16[(size_t)m * RNN_H + nc] = (_Float16)v;
}

// =====================================================================
// Wo [1024 x 32000] fp32  ->  WoT [32000 x 1024] f16  (tiled via LDS)
// Grid: dim3(16, 500), 256 threads; 64x64 tiles, both phases coalesced.
// =====================================================================
__global__ __launch_bounds__(256) void cvt_transpose_wo(
    const float* __restrict__ Wo, _Float16* __restrict__ WoT)
{
    __shared__ _Float16 tile[64][72];   // padded rows to dodge bank conflicts
    const int kb = blockIdx.x * 64;
    const int nb = blockIdx.y * 64;

    #pragma unroll 4
    for (int p = 0; p < 16; ++p) {
        const int row = p * 4 + (threadIdx.x >> 6);   // k within tile
        const int col = threadIdx.x & 63;             // n within tile (coalesced)
        tile[col][row] = (_Float16)Wo[(size_t)(kb + row) * RNN_V + nb + col];
    }
    __syncthreads();
    #pragma unroll 4
    for (int p = 0; p < 16; ++p) {
        const int nrow = p * 4 + (threadIdx.x >> 6);  // n within tile
        const int kcol = threadIdx.x & 63;            // k within tile (coalesced)
        WoT[(size_t)(nb + nrow) * RNN_H + kb + kcol] = tile[nrow][kcol];
    }
}

// =====================================================================
// Output projection: out[b,t,v] = h2s[t,b,:] . Wo[:,v] + bo[v]
// A = h2s_f16 [T*B x H], B = WoT f16 [V x H] (K-major per row).
// Block: 128(M) x 128(N) tile, Kb=64, DOUBLE-BUFFERED LDS panels filled
// with GLOBAL_LOAD_ASYNC_TO_LDS_B128 (ASYNCcnt) so Wo streaming overlaps
// the V_WMMA_F32_16X16X32_F16 inner loop. 8 waves x (2Mx4N) 16x16 tiles.
// Grid: 16 * 250 = 4000 blocks, 256 threads.
// =====================================================================
__global__ __launch_bounds__(256) void out_proj_gemm(
    const _Float16* __restrict__ h2s, const _Float16* __restrict__ WoT,
    const float* __restrict__ bo, float* __restrict__ out)
{
    constexpr int SK    = 80;   // LDS row stride (halves); 160B, 16B-aligned
    constexpr int KB    = 64;   // K-panel depth
    constexpr int NITER = RNN_H / KB;   // 16 pipeline stages

    __shared__ __align__(16) _Float16 As[2][128 * SK];
    __shared__ __align__(16) _Float16 Bs[2][128 * SK];

    const int mb   = blockIdx.x & 15;            // 16 M-blocks (128 rows each)
    const int nb   = blockIdx.x >> 4;            // 250 N-blocks
    const int tid  = threadIdx.x;
    const int lane = tid & 31;
    const int w    = tid >> 5;
    const int mw   = w & 3;                      // wave's M quadrant (2 subtiles)
    const int nw   = w >> 2;                     // wave's N half     (4 subtiles)
    const int l16  = lane & 15;
    const int s    = lane >> 4;

    // per-thread staging coordinates: thread covers 64B (4x b128) of one row
    const int srow  = tid >> 1;
    const int shalf = tid & 1;
    const unsigned long long gA0 =
        (unsigned long long)(h2s + (size_t)(mb * 128 + srow) * RNN_H + shalf * 32);
    const unsigned long long gB0 =
        (unsigned long long)(WoT + (size_t)(nb * 128 + srow) * RNN_H + shalf * 32);
    const unsigned int lA0 = lds_offset_of(&As[0][srow * SK + shalf * 32]);
    const unsigned int lB0 = lds_offset_of(&Bs[0][srow * SK + shalf * 32]);
    const unsigned int bufStride = lds_offset_of(&As[1][0]) - lds_offset_of(&As[0][0]);

    auto issue_stage = [&](int kb, int buf) {
        const unsigned long long ga = gA0 + (size_t)kb * sizeof(_Float16);
        const unsigned long long gb = gB0 + (size_t)kb * sizeof(_Float16);
        const unsigned int la = lA0 + buf * bufStride;
        const unsigned int lb = lB0 + buf * bufStride;
        #pragma unroll
        for (int j = 0; j < 4; ++j) {            // 8 async ops / thread / stage
            async_copy_b128(ga + j * 16, la + j * 16);
            async_copy_b128(gb + j * 16, lb + j * 16);
        }
    };

    v8f acc[2][4] = {};

    issue_stage(0, 0);                            // prime the pipeline
    for (int it = 0; it < NITER; ++it) {
        const int buf = it & 1;
        if (it + 1 < NITER) {
            // buf^1 was last read in compute(it-1); trailing barrier below
            // guarantees every wave is done with it before this overwrite.
            issue_stage((it + 1) * KB, buf ^ 1);
            wait_async_le8();                     // my stage-it loads landed
        } else {
            wait_async_le0();
        }
        __syncthreads();                          // all waves' stage-it data visible

        #pragma unroll
        for (int ks = 0; ks < KB; ks += 32) {
            union Frag { uint4 u[2]; v16h h; } a[2], b[4];
            #pragma unroll
            for (int mi = 0; mi < 2; ++mi) {
                const int mloc = (2 * mw + mi) * 16 + l16;
                const uint4* p = (const uint4*)(&As[buf][mloc * SK + ks + s * 8]);
                a[mi].u[0] = p[0];                // K = ks + s*8 .. +7
                a[mi].u[1] = p[2];                // K = ks + 16 + s*8 .. +7
            }
            #pragma unroll
            for (int ni = 0; ni < 4; ++ni) {
                const int nloc = (4 * nw + ni) * 16 + l16;
                const uint4* p = (const uint4*)(&Bs[buf][nloc * SK + ks + s * 8]);
                b[ni].u[0] = p[0];
                b[ni].u[1] = p[2];
            }
            #pragma unroll
            for (int mi = 0; mi < 2; ++mi)
                #pragma unroll
                for (int ni = 0; ni < 4; ++ni)
                    acc[mi][ni] = __builtin_amdgcn_wmma_f32_16x16x32_f16(
                        false, a[mi].h, false, b[ni].h, (short)0,
                        acc[mi][ni], false, false);
        }
        __syncthreads();                          // compute(it) done before buf reuse
    }

    // ---- epilogue: each 16-row M-subtile is exactly one timestep t ----
    #pragma unroll
    for (int mi = 0; mi < 2; ++mi) {
        const int tg = mb * 8 + 2 * mw + mi;                 // global t
        #pragma unroll
        for (int ni = 0; ni < 4; ++ni) {
            const int v  = nb * 128 + (4 * nw + ni) * 16 + l16;
            const float bv = bo[v];
            #pragma unroll
            for (int i = 0; i < 8; ++i) {
                const int brow = i + s * 8;                  // global b
                out[(size_t)brow * RNN_T * RNN_V + (size_t)tg * RNN_V + v]
                    = acc[mi][ni][i] + bv;
            }
        }
    }
}

// =====================================================================
// Host-side orchestration (graph-capture safe: only async ops on stream)
// =====================================================================
extern "C" void kernel_launch(void* const* d_in, const int* in_sizes, int n_in,
                              void* d_out, int out_size, void* d_ws, size_t ws_size,
                              hipStream_t stream)
{
    (void)in_sizes; (void)n_in; (void)out_size; (void)ws_size;

    const float* x   = (const float*)d_in[0];
    const float* Wx1 = (const float*)d_in[1];
    const float* Wh1 = (const float*)d_in[2];
    const float* b1  = (const float*)d_in[3];
    const float* Wx2 = (const float*)d_in[4];
    const float* Wh2 = (const float*)d_in[5];
    const float* b2  = (const float*)d_in[6];
    const float* Wo  = (const float*)d_in[7];
    const float* bo  = (const float*)d_in[8];
    float* out = (float*)d_out;

    // ---- workspace carve-up ----
    char* ws = (char*)d_ws;
    size_t off = 0;
    auto carve = [&](size_t bytes) { char* p = ws + off; off = (off + bytes + 255) & ~(size_t)255; return p; };
    float*    xproj = (float*)   carve(RNN_B * RNN_H * sizeof(float));
    float*    h1a   = (float*)   carve(RNN_B * RNN_H * sizeof(float));
    float*    h1b   = (float*)   carve(RNN_B * RNN_H * sizeof(float));
    float*    h2a   = (float*)   carve(RNN_B * RNN_H * sizeof(float));
    float*    h2b   = (float*)   carve(RNN_B * RNN_H * sizeof(float));
    _Float16* h2s16 = (_Float16*)carve((size_t)RNN_T * RNN_B * RNN_H * sizeof(_Float16));
    _Float16* WoT   = (_Float16*)carve((size_t)RNN_V * RNN_H * sizeof(_Float16));

    float* h1[2] = { h1a, h1b };
    float* h2[2] = { h2a, h2b };

    // zero-init initial hidden states (graph-capturable async memsets)
    hipMemsetAsync(h1a, 0, RNN_B * RNN_H * sizeof(float), stream);
    hipMemsetAsync(h2a, 0, RNN_B * RNN_H * sizeof(float), stream);

    // Wo -> WoT (f16, transposed, K-contiguous per output column)
    cvt_transpose_wo<<<dim3(RNN_H / 64, RNN_V / 64), 256, 0, stream>>>(Wo, WoT);

    // xproj = x @ Wx1 + b1  (hoisted, step-invariant)
    rnn_step_gemm<<<RNN_H / 16, 256, 0, stream>>>(
        x, Wx1, nullptr, nullptr, b1, nullptr, xproj, nullptr, 0);

    // sequential recurrence
    for (int t = 0; t < RNN_T; ++t) {
        const int cur = t & 1, nxt = cur ^ 1;
        // h1' = tanh(xproj + h1 @ Wh1)
        rnn_step_gemm<<<RNN_H / 16, 256, 0, stream>>>(
            h1[cur], Wh1, nullptr, nullptr, nullptr, xproj, h1[nxt], nullptr, 1);
        // h2' = tanh(h1' @ Wx2 + h2 @ Wh2 + b2); also emit f16 copy into h2s[t]
        rnn_step_gemm<<<RNN_H / 16, 256, 0, stream>>>(
            h1[nxt], Wx2, h2[cur], Wh2, b2, nullptr, h2[nxt],
            h2s16 + (size_t)t * RNN_B * RNN_H, 1);
    }

    // logits[b,t,v] = h2s . Wo + bo
    out_proj_gemm<<<(RNN_T * RNN_B / 128) * (RNN_V / 128), 256, 0, stream>>>(
        h2s16, WoT, bo, out);
}
